// GraphVAE_78941498901078
// MI455X (gfx1250) — matrix operand order repaired and checked by reading, not compiled, not run on hardware
//
#include <hip/hip_runtime.h>
#include <hip/hip_bf16.h>
#include <stdint.h>

#define N_NODES 100000
#define N_EDGES 1600000
#define N_GRAPHS 512
#define IN_CH 64
#define HID 128
#define LAT 64
#define MAXN 128
#define OUT_P 8128    // MAXN*(MAXN-1)/2
#define OUT_PAD 8192  // OUT_P padded to a multiple of 128 (branch-free GEMM)

typedef __attribute__((ext_vector_type(16))) __bf16    v16bf;
typedef __attribute__((ext_vector_type(8)))  float     v8f;
typedef __attribute__((ext_vector_type(4)))  uint32_t  u32x4;

union BF16Frag { v16bf v; uint32_t u[8]; u32x4 q[2]; };

__device__ __forceinline__ uint16_t f2bf(float f) {
    uint32_t u = __float_as_uint(f);
    uint32_t r = u + 0x7FFFu + ((u >> 16) & 1u);   // round-to-nearest-even
    return (uint16_t)(r >> 16);
}

// ---------------- utility kernels ----------------

__global__ void zero_k(float* p, size_t n) {
    size_t t = (size_t)blockIdx.x * blockDim.x + threadIdx.x;
    size_t stride = (size_t)gridDim.x * blockDim.x;
    for (; t < n; t += stride) p[t] = 0.0f;
}

__global__ void cvt_bf16_k(const float* __restrict__ in, uint16_t* __restrict__ out, int n) {
    int t = blockIdx.x * blockDim.x + threadIdx.x;
    if (t < n) out[t] = f2bf(in[t]);
}

// pack fp32 weight [K,N] into bf16 WMMA-B fragment order over padded width Npad:
// dword index = ((nt*KB + kb)*32 + lane)*8 + v ; pair (k,k+1), k = kb*32 + 2v + 8*(lane>>4) + 8*(v>>2)
// columns n >= N are zero-filled.
__global__ void pack_w_k(const float* __restrict__ W, uint32_t* __restrict__ Wp,
                         int K, int N, int Npad) {
    int t = blockIdx.x * blockDim.x + threadIdx.x;
    if (t >= (K * Npad) / 2) return;
    int v    = t & 7;
    int lane = (t >> 3) & 31;
    int KB   = K >> 5;
    int kb   = (t >> 8) % KB;
    int nt   = (t >> 8) / KB;
    int n    = nt * 16 + (lane & 15);
    int k    = kb * 32 + 2 * v + 8 * (lane >> 4) + 8 * (v >> 2);
    uint32_t val = 0;
    if (n < N) {
        val = (uint32_t)f2bf(W[(size_t)k * N + n]) |
              ((uint32_t)f2bf(W[(size_t)(k + 1) * N + n]) << 16);
    }
    Wp[t] = val;
}

__global__ void degree_k(const int* __restrict__ dst, float* __restrict__ deg, int ne) {
    int e = blockIdx.x * blockDim.x + threadIdx.x;
    if (e < ne) atomicAdd(&deg[dst[e]], 1.0f);
}

__global__ void dinv_k(float* __restrict__ deg, int n) {
    int i = blockIdx.x * blockDim.x + threadIdx.x;
    if (i < n) deg[i] = rsqrtf(deg[i] + 1.0f);   // in-place: deg -> d^{-1/2}
}

// ---------------- bf16 WMMA GEMM ----------------
// A: bf16 row-major [M,K] (M % 16 == 0); Wp: fragment-packed bf16 [K,Npad] (Npad % 128 == 0).
// C = act(A@W + bias), stored over [M,Npad] with columns >= Nvalid skipped (tile-uniform).
// block = 128 threads (4 waves); each wave owns TWO 16x16 tiles sharing one A fragment.
// K-loop fully unrolled; no divergent control flow anywhere near the WMMAs.
template <int K, int ACT, bool BF16OUT>
__global__ void gemm_wmma(const uint16_t* __restrict__ Ab, const uint32_t* __restrict__ Wp,
                          const float* __restrict__ bias, void* __restrict__ Cv,
                          int Npad, int Nvalid) {
    constexpr int KB = K / 32;
    const int wave = threadIdx.x >> 5;
    const int lane = threadIdx.x & 31;
    const int m0 = blockIdx.x * 16;
    const int n0 = blockIdx.y * 128 + wave * 32;   // two 16-col tiles per wave
    const int h  = lane >> 4;
    const int mr = lane & 15;

    const uint16_t* Arow = Ab + (size_t)(m0 + mr) * K;
    const int nt0 = n0 >> 4;
    const uint32_t* Wp0 = Wp + (size_t)nt0 * KB * 256;        // 256 dwords per (nt,kb)
    const uint32_t* Wp1 = Wp + (size_t)(nt0 + 1) * KB * 256;

    v8f acc0 = {}, acc1 = {};
#pragma unroll
    for (int kb = 0; kb < KB; ++kb) {
        const int k0 = kb * 32;
        BF16Frag a, b0, b1;
        // A fragment: two contiguous 16B runs per lane
        a.q[0] = *(const u32x4*)(const void*)(Arow + k0 + 8 * h);
        a.q[1] = *(const u32x4*)(const void*)(Arow + k0 + 16 + 8 * h);
        // B fragments: two b128 loads each, fully coalesced
        const uint32_t* bp0 = Wp0 + kb * 256 + lane * 8;
        const uint32_t* bp1 = Wp1 + kb * 256 + lane * 8;
        b0.q[0] = *(const u32x4*)(const void*)(bp0);
        b0.q[1] = *(const u32x4*)(const void*)(bp0 + 4);
        b1.q[0] = *(const u32x4*)(const void*)(bp1);
        b1.q[1] = *(const u32x4*)(const void*)(bp1 + 4);
        acc0 = __builtin_amdgcn_wmma_f32_16x16x32_bf16(false, a.v, false, b0.v,
                                                       (short)0, acc0, false, false);
        acc1 = __builtin_amdgcn_wmma_f32_16x16x32_bf16(false, a.v, false, b1.v,
                                                       (short)0, acc1, false, false);
    }

    auto store_tile = [&](const v8f& acc, int nbase) {
        int col = nbase + mr;
        float bv = bias ? bias[col] : 0.0f;
#pragma unroll
        for (int r = 0; r < 8; ++r) {
            int orow = m0 + r + 8 * h;
            float val = acc[r] + bv;
            if constexpr (ACT == 1) val = fmaxf(val, 0.0f);
            else if constexpr (ACT == 2) val = 1.0f / (1.0f + __expf(-val));
            if constexpr (BF16OUT) ((uint16_t*)Cv)[(size_t)orow * Npad + col] = f2bf(val);
            else                   ((float*)Cv)[(size_t)orow * Npad + col] = val;
        }
    };
    if (n0 < Nvalid)      store_tile(acc0, n0);        // tile-uniform guards,
    if (n0 + 16 < Nvalid) store_tile(acc1, n0 + 16);   // outside the WMMA region
}

// ---------------- GCN edge scatter: agg[dst] += h[src] * dinv[src]*dinv[dst] ----------------
__global__ void scatter_k(const float* __restrict__ h, const int* __restrict__ src,
                          const int* __restrict__ dst, const float* __restrict__ dinv,
                          float* __restrict__ agg, int ne, int C) {
    int t = blockIdx.x * blockDim.x + threadIdx.x;
    int e = t >> 5, lane = t & 31;
    if (e >= ne) return;
    int s = src[e], d = dst[e];
    float norm = dinv[s] * dinv[d];
    const float* hs = h + (size_t)s * C;
    float* ad = agg + (size_t)d * C;
#pragma unroll
    for (int c = lane; c < C; c += 32)
        atomicAdd(&ad[c], hs[c] * norm);
}

// out = relu(agg + hw * dinv^2 + b); output fp32 or bf16 (when it only feeds a GEMM)
template <bool BF16OUT>
__global__ void combine_k(const float* __restrict__ hw, const float* __restrict__ agg,
                          const float* __restrict__ dinv, const float* __restrict__ bias,
                          void* __restrict__ out, int n, int C) {
    size_t t = (size_t)blockIdx.x * blockDim.x + threadIdx.x;
    if (t >= (size_t)n * C) return;
    int i = (int)(t / C), c = (int)(t % C);
    float di = dinv[i];
    float v = fmaxf(agg[t] + hw[t] * di * di + bias[c], 0.0f);
    if constexpr (BF16OUT) ((uint16_t*)out)[t] = f2bf(v);
    else                   ((float*)out)[t] = v;
}

__global__ void pool_k(const float* __restrict__ h, const int* __restrict__ batch,
                       float* __restrict__ gsum, float* __restrict__ gcnt, int n, int C) {
    size_t t = (size_t)blockIdx.x * blockDim.x + threadIdx.x;
    if (t >= (size_t)n * C) return;
    int i = (int)(t / C), c = (int)(t % C);
    int g = batch[i];
    atomicAdd(&gsum[(size_t)g * C + c], h[t]);
    if (c == 0) atomicAdd(&gcnt[g], 1.0f);
}

// mu/logvar to d_out (fp32), z in bf16 (feeds decoder GEMM only)
__global__ void latent_k(const float* __restrict__ gsum, const float* __restrict__ gcnt,
                         const float* __restrict__ Wmu, const float* __restrict__ bmu,
                         const float* __restrict__ Wlv, const float* __restrict__ blv,
                         const float* __restrict__ eps, float* __restrict__ mu_out,
                         float* __restrict__ lv_out, uint16_t* __restrict__ zb) {
    int t = blockIdx.x * blockDim.x + threadIdx.x;
    if (t >= N_GRAPHS * LAT) return;
    int g = t / LAT, l = t % LAT;
    float cnt = fmaxf(gcnt[g], 1.0f);
    float mu = bmu[l], lv = blv[l];
    const float* gs = gsum + (size_t)g * HID;
#pragma unroll 4
    for (int c = 0; c < HID; ++c) {
        float hm = gs[c] / cnt;
        mu += hm * Wmu[c * LAT + l];
        lv += hm * Wlv[c * LAT + l];
    }
    mu_out[t] = mu;
    lv_out[t] = lv;
    zb[t] = f2bf(mu + eps[t] * __expf(0.5f * lv));
}

// triu(k=1) index tables for MAXN
__global__ void triu_k(int* __restrict__ rtab, int* __restrict__ ctab) {
    int r = blockIdx.x * blockDim.x + threadIdx.x;
    if (r >= MAXN - 1) return;
    int off = r * (MAXN - 1) - r * (r - 1) / 2;
    for (int c = r + 1; c < MAXN; ++c) { rtab[off] = r; ctab[off] = c; ++off; }
}

__global__ void adj_k(const float* __restrict__ probs, const int* __restrict__ rtab,
                      const int* __restrict__ ctab, float* __restrict__ adj) {
    int t = blockIdx.x * blockDim.x + threadIdx.x;
    if (t >= N_GRAPHS * OUT_P) return;
    int g = t / OUT_P, p = t % OUT_P;
    float v = probs[(size_t)g * OUT_PAD + p];      // padded row stride
    int r = rtab[p], c = ctab[p];
    adj[((size_t)g * MAXN + r) * MAXN + c] = v;
    adj[((size_t)g * MAXN + c) * MAXN + r] = v;
}

__global__ void diag_k(float* __restrict__ adj) {
    int t = blockIdx.x * blockDim.x + threadIdx.x;
    if (t >= N_GRAPHS * MAXN) return;
    int g = t / MAXN, i = t % MAXN;
    adj[((size_t)g * MAXN + i) * MAXN + i] = 0.0f;
}

// ---------------- host ----------------

extern "C" void kernel_launch(void* const* d_in, const int* in_sizes, int n_in,
                              void* d_out, int out_size, void* d_ws, size_t ws_size,
                              hipStream_t stream) {
    (void)in_sizes; (void)n_in; (void)out_size; (void)ws_size;
    const float* x    = (const float*)d_in[0];
    const int*   ei   = (const int*)d_in[1];
    const int*   batch= (const int*)d_in[2];
    const float* eps  = (const float*)d_in[3];
    const float* W1   = (const float*)d_in[4];
    const float* b1   = (const float*)d_in[5];
    const float* W2   = (const float*)d_in[6];
    const float* b2   = (const float*)d_in[7];
    const float* Wmu  = (const float*)d_in[8];
    const float* bmu  = (const float*)d_in[9];
    const float* Wlv  = (const float*)d_in[10];
    const float* blv  = (const float*)d_in[11];
    const float* D1   = (const float*)d_in[12];
    const float* d1   = (const float*)d_in[13];
    const float* D2   = (const float*)d_in[14];
    const float* d2   = (const float*)d_in[15];
    const float* D3   = (const float*)d_in[16];
    const float* d3   = (const float*)d_in[17];

    const int* src = ei;
    const int* dst = ei + N_EDGES;

    float* out = (float*)d_out;
    float* adj    = out;                                    // [512,128,128]
    float* mu_out = out + (size_t)N_GRAPHS * MAXN * MAXN;
    float* lv_out = mu_out + (size_t)N_GRAPHS * LAT;

    // workspace layout (in float units)
    float* wf = (float*)d_ws;
    size_t o = 0;
    float* dinv = wf + o; o += 100352;
    const size_t NH = (size_t)N_NODES * HID;
    float* buf0 = wf + o; o += NH;
    float* buf1 = wf + o; o += NH;
    uint16_t* h1b = (uint16_t*)(wf + o); o += NH / 2;       // bf16 h1
    float* gsum = wf + o; o += (size_t)N_GRAPHS * HID;
    float* gcnt = wf + o; o += 512;
    uint32_t* w1p = (uint32_t*)(wf + o); o += (IN_CH * HID) / 2;
    uint32_t* w2p = (uint32_t*)(wf + o); o += (HID * HID) / 2;
    uint32_t* d1p = (uint32_t*)(wf + o); o += (LAT * HID) / 2;
    uint32_t* d2p = (uint32_t*)(wf + o); o += (HID * HID) / 2;
    uint32_t* d3p = (uint32_t*)(wf + o); o += (HID * OUT_PAD) / 2;
    int* rtab = (int*)(wf + o); o += OUT_P;
    int* ctab = (int*)(wf + o); o += OUT_P;
    uint16_t* zb  = (uint16_t*)(wf + o); o += (N_GRAPHS * LAT) / 2;
    uint16_t* p1b = (uint16_t*)(wf + o); o += (N_GRAPHS * HID) / 2;
    uint16_t* p2b = (uint16_t*)(wf + o); o += (N_GRAPHS * HID) / 2;
    // aliases into big buffers (lifetimes don't overlap):
    uint16_t* xb    = (uint16_t*)buf1;                      // bf16 x; dead before buf1 is zeroed
    float*    probs = buf0;                                 // [512,OUT_PAD]; buf0 free after combine2

    const int T = 256;
    auto blocks = [](size_t n, int t) { return (unsigned)((n + t - 1) / t); };

    // degrees -> dinv
    zero_k<<<blocks(N_NODES, T), T, 0, stream>>>(dinv, N_NODES);
    degree_k<<<blocks(N_EDGES, T), T, 0, stream>>>(dst, dinv, N_EDGES);
    dinv_k<<<blocks(N_NODES, T), T, 0, stream>>>(dinv, N_NODES);

    // pack weights into WMMA fragment order (bf16), zero-padding D3 to OUT_PAD
    pack_w_k<<<blocks(IN_CH * HID / 2, T), T, 0, stream>>>(W1, w1p, IN_CH, HID, HID);
    pack_w_k<<<blocks(HID * HID / 2, T), T, 0, stream>>>(W2, w2p, HID, HID, HID);
    pack_w_k<<<blocks(LAT * HID / 2, T), T, 0, stream>>>(D1, d1p, LAT, HID, HID);
    pack_w_k<<<blocks(HID * HID / 2, T), T, 0, stream>>>(D2, d2p, HID, HID, HID);
    pack_w_k<<<blocks((size_t)HID * OUT_PAD / 2, T), T, 0, stream>>>(D3, d3p, HID, OUT_P, OUT_PAD);

    // x -> bf16
    cvt_bf16_k<<<blocks((size_t)N_NODES * IN_CH, T), T, 0, stream>>>(x, xb, N_NODES * IN_CH);

    // ---- GCN layer 1 ----
    gemm_wmma<IN_CH, 0, false><<<dim3(N_NODES / 16, HID / 128), 128, 0, stream>>>(
        xb, w1p, nullptr, buf0, HID, HID);                  // hw1 -> buf0 (f32)
    zero_k<<<blocks(NH, T), T, 0, stream>>>(buf1, NH);
    scatter_k<<<blocks((size_t)N_EDGES * 32, T), T, 0, stream>>>(buf0, src, dst, dinv, buf1, N_EDGES, HID);
    combine_k<true><<<blocks(NH, T), T, 0, stream>>>(buf0, buf1, dinv, b1, h1b, N_NODES, HID);

    // ---- GCN layer 2 ----
    gemm_wmma<HID, 0, false><<<dim3(N_NODES / 16, HID / 128), 128, 0, stream>>>(
        h1b, w2p, nullptr, buf1, HID, HID);                 // hw2 -> buf1 (f32)
    zero_k<<<blocks(NH, T), T, 0, stream>>>(buf0, NH);
    scatter_k<<<blocks((size_t)N_EDGES * 32, T), T, 0, stream>>>(buf1, src, dst, dinv, buf0, N_EDGES, HID);
    combine_k<false><<<blocks(NH, T), T, 0, stream>>>(buf1, buf0, dinv, b2, buf1, N_NODES, HID);

    // ---- global mean pool ----
    zero_k<<<blocks((size_t)N_GRAPHS * HID + 512, T), T, 0, stream>>>(gsum, (size_t)N_GRAPHS * HID + 512);
    pool_k<<<blocks(NH, T), T, 0, stream>>>(buf1, batch, gsum, gcnt, N_NODES, HID);

    // ---- latent ----
    latent_k<<<blocks(N_GRAPHS * LAT, T), T, 0, stream>>>(gsum, gcnt, Wmu, bmu, Wlv, blv,
                                                          eps, mu_out, lv_out, zb);

    // ---- decoder MLP (WMMA, fused bias+activation, bf16 chaining) ----
    gemm_wmma<LAT, 1, true><<<dim3(N_GRAPHS / 16, HID / 128), 128, 0, stream>>>(
        zb, d1p, d1, p1b, HID, HID);                        // relu, bf16 out
    gemm_wmma<HID, 1, true><<<dim3(N_GRAPHS / 16, HID / 128), 128, 0, stream>>>(
        p1b, d2p, d2, p2b, HID, HID);                       // relu, bf16 out
    gemm_wmma<HID, 2, false><<<dim3(N_GRAPHS / 16, OUT_PAD / 128), 128, 0, stream>>>(
        p2b, d3p, d3, probs, OUT_PAD, OUT_P);               // sigmoid, f32 out (padded N)

    // ---- symmetric adjacency scatter ----
    triu_k<<<1, 128, 0, stream>>>(rtab, ctab);
    adj_k<<<blocks((size_t)N_GRAPHS * OUT_P, T), T, 0, stream>>>(probs, rtab, ctab, adj);
    diag_k<<<blocks(N_GRAPHS * MAXN, T), T, 0, stream>>>(adj);
}
